// ConditionalHeteroGraphVAE_22823456211242
// MI455X (gfx1250) — matrix operand order repaired and checked by reading, not compiled, not run on hardware
//
#include <hip/hip_runtime.h>
#include <hip/hip_bf16.h>

typedef __attribute__((ext_vector_type(16))) _Float16 v16h;
typedef __attribute__((ext_vector_type(8)))  float    v8f;

#define HIDF 64

// ---- WMMA VGPR index maps (CDNA5 ISA §7.12.2) ----
// A 16x32 f16: lanes 0-15 hold K {0..7,16..23}, lanes 16-31 hold K {8..15,24..31}
// B 32x16 f16: lanes 0-15 hold K {0..7,16..23}, lanes 16-31 hold K {8..15,24..31}
__device__ __forceinline__ int b_kmap(int i, int g) { return (i < 8) ? 8 * g + i : 16 + 8 * g + (i - 8); }

__device__ __forceinline__ void atomicMaxF(float* addr, float val) {
  if (val >= 0.f) atomicMax((int*)addr, __float_as_int(val));
  else            atomicMin((unsigned int*)addr, __float_as_uint(val));
}

// branchless tanh: 1 - 2/(exp(2x)+1); saturates to +-1 for large |x|
__device__ __forceinline__ float fast_tanh(float x) {
  return 1.f - __fdividef(2.f, __expf(2.f * x) + 1.f);
}

__device__ __forceinline__ void cvt8(v16h& a, int base, const float4& lo, const float4& hi) {
  a[base + 0] = (_Float16)lo.x; a[base + 1] = (_Float16)lo.y;
  a[base + 2] = (_Float16)lo.z; a[base + 3] = (_Float16)lo.w;
  a[base + 4] = (_Float16)hi.x; a[base + 5] = (_Float16)hi.y;
  a[base + 6] = (_Float16)hi.z; a[base + 7] = (_Float16)hi.w;
}
__device__ __forceinline__ void cvt8r(v16h& a, int base, const float4& lo, const float4& hi) {
  a[base + 0] = (_Float16)fmaxf(lo.x, 0.f); a[base + 1] = (_Float16)fmaxf(lo.y, 0.f);
  a[base + 2] = (_Float16)fmaxf(lo.z, 0.f); a[base + 3] = (_Float16)fmaxf(lo.w, 0.f);
  a[base + 4] = (_Float16)fmaxf(hi.x, 0.f); a[base + 5] = (_Float16)fmaxf(hi.y, 0.f);
  a[base + 6] = (_Float16)fmaxf(hi.z, 0.f); a[base + 7] = (_Float16)fmaxf(hi.w, 0.f);
}

// =============== one-shot B-fragment packers (f16, WMMA lane layout, zero-padded) ===============
__global__ void pack_B_kernel(const float* __restrict__ W, _Float16* __restrict__ pack, int F) {
  int t = threadIdx.x;              // 0..127 = nt*32 + lane
  int nt = t >> 5, lane = t & 31;
  int g = lane >> 4, m = lane & 15;
  int n = nt * 16 + m;
  for (int i = 0; i < 16; ++i) {
    int k = b_kmap(i, g);
    pack[(size_t)t * 16 + i] = (_Float16)((k < F) ? W[k * HIDF + n] : 0.f);
  }
}
__global__ void pack_Wk_kernel(const float* __restrict__ Wk, _Float16* __restrict__ pack) {
  int t = threadIdx.x;              // 0..255 = (ks*4+nt)*32 + lane
  int lane = t & 31, frag = t >> 5;
  int ks = frag >> 2, nt = frag & 3;
  int g = lane >> 4, m = lane & 15;
  int n = nt * 16 + m;
  for (int i = 0; i < 16; ++i) {
    int k = 32 * ks + b_kmap(i, g);
    pack[(size_t)t * 16 + i] = (_Float16)Wk[k * HIDF + n];
  }
}

// =============== projection: Hout[n,64] = X[n,F] @ W[F,64] + bias ===============
template <int F>
__global__ void proj_wmma_kernel(const float* __restrict__ X,
                                 const _Float16* __restrict__ Bpack,
                                 const float* __restrict__ bias,
                                 float* __restrict__ Hout) {
  const int tile = blockIdx.x;
  const int lane = threadIdx.x;
  const int g = lane >> 4, m = lane & 15;
  const int row = tile * 16 + m;
  v16h a;
  if (F == 16) {
    float4 x0 = *(const float4*)(X + (size_t)row * 16 + 8 * g);
    float4 x1 = *(const float4*)(X + (size_t)row * 16 + 8 * g + 4);
    cvt8(a, 0, x0, x1);
  } else {
    float4 x0 = *(const float4*)(X + (size_t)row * 8);
    float4 x1 = *(const float4*)(X + (size_t)row * 8 + 4);
    float sel = g ? 0.f : 1.f;
    x0.x *= sel; x0.y *= sel; x0.z *= sel; x0.w *= sel;
    x1.x *= sel; x1.y *= sel; x1.z *= sel; x1.w *= sel;
    cvt8(a, 0, x0, x1);
  }
#pragma unroll
  for (int i = 8; i < 16; ++i) a[i] = (_Float16)0.f;

#pragma unroll
  for (int nt = 0; nt < 4; ++nt) {
    const int n = nt * 16 + m;
    v16h b = *(const v16h*)(Bpack + (size_t)(nt * 32 + lane) * 16);
    v8f c;
#pragma unroll
    for (int r = 0; r < 8; ++r) c[r] = bias[n];
    c = __builtin_amdgcn_wmma_f32_16x16x32_f16(false, a, false, b, (short)0, c, false, false);
#pragma unroll
    for (int r = 0; r < 8; ++r)
      Hout[(size_t)(tile * 16 + 8 * g + r) * HIDF + n] = c[r];
  }
}

// =============== per-node attention logits ===============
__global__ void node_scores_kernel(const float* __restrict__ Hn,
                                   const float* __restrict__ att,
                                   float* __restrict__ s, int n4) {
  int idx = blockIdx.x * blockDim.x + threadIdx.x;
  if (idx >= n4) return;
  int node = idx >> 2, h = idx & 3;
  const float4* hp = (const float4*)(Hn + (size_t)node * HIDF + h * 16);
  const float4* ap = (const float4*)(att + h * 16);
  float acc = 0.f;
#pragma unroll
  for (int v = 0; v < 4; ++v) {
    float4 hv = hp[v], av = ap[v];
    acc += hv.x * av.x + hv.y * av.y + hv.z * av.z + hv.w * av.w;
  }
  s[idx] = acc;
}

// =============== per-type init ===============
__global__ void seg_init_kernel(float* __restrict__ o, float* __restrict__ amax,
                                float* __restrict__ denom, int nd) {
  int i = blockIdx.x * blockDim.x + threadIdx.x;
  if (i < nd * HIDF) o[i] = 0.f;
  if (i < nd * 4) { amax[i] = -__builtin_huge_valf(); denom[i] = 0.f; }
}

__global__ void zero_kernel(float* __restrict__ p, int n) {
  int i = blockIdx.x * blockDim.x + threadIdx.x;
  if (i < n) p[i] = 0.f;
}

__device__ __forceinline__ float lrelu02(float a) { return a > 0.f ? a : 0.2f * a; }

// =============== pass 1: segment max ===============
__global__ void edge_amax_kernel(const int* __restrict__ ei, int E,
                                 const float* __restrict__ ss,
                                 const float* __restrict__ sd,
                                 float* __restrict__ amax) {
  int e = blockIdx.x * blockDim.x + threadIdx.x;
  if (e >= E) return;
  int s = ei[e], d = ei[E + e];
  float4 vs = *(const float4*)(ss + (size_t)s * 4);
  float4 vd = *(const float4*)(sd + (size_t)d * 4);
  atomicMaxF(&amax[d * 4 + 0], lrelu02(vs.x + vd.x));
  atomicMaxF(&amax[d * 4 + 1], lrelu02(vs.y + vd.y));
  atomicMaxF(&amax[d * 4 + 2], lrelu02(vs.z + vd.z));
  atomicMaxF(&amax[d * 4 + 3], lrelu02(vs.w + vd.w));
}

// =============== pass 2: segment sum of exp(a - amax) ===============
__global__ void edge_exp_kernel(const int* __restrict__ ei, int E,
                                const float* __restrict__ ss,
                                const float* __restrict__ sd,
                                const float* __restrict__ amax,
                                float* __restrict__ denom) {
  int e = blockIdx.x * blockDim.x + threadIdx.x;
  if (e >= E) return;
  int s = ei[e], d = ei[E + e];
  float4 vs = *(const float4*)(ss + (size_t)s * 4);
  float4 vd = *(const float4*)(sd + (size_t)d * 4);
  float4 vm = *(const float4*)(amax + (size_t)d * 4);
  atomicAdd(&denom[d * 4 + 0], __expf(lrelu02(vs.x + vd.x) - vm.x));
  atomicAdd(&denom[d * 4 + 1], __expf(lrelu02(vs.y + vd.y) - vm.y));
  atomicAdd(&denom[d * 4 + 2], __expf(lrelu02(vs.z + vd.z) - vm.z));
  atomicAdd(&denom[d * 4 + 3], __expf(lrelu02(vs.w + vd.w) - vm.w));
}

// =============== pass 3: o[dst] += h_src * alpha ===============
__global__ void edge_msg_kernel(const int* __restrict__ ei, int E,
                                const float* __restrict__ ss,
                                const float* __restrict__ sd,
                                const float* __restrict__ amax,
                                const float* __restrict__ denom,
                                const float* __restrict__ Hsrc,
                                float* __restrict__ o) {
  int e = blockIdx.x * blockDim.x + threadIdx.x;
  if (e >= E) return;
  int s = ei[e], d = ei[E + e];
  float4 vs = *(const float4*)(ss + (size_t)s * 4);
  float4 vd = *(const float4*)(sd + (size_t)d * 4);
  float4 vm = *(const float4*)(amax + (size_t)d * 4);
  float4 vq = *(const float4*)(denom + (size_t)d * 4);
  float alpha[4];
  alpha[0] = __expf(lrelu02(vs.x + vd.x) - vm.x) / (vq.x + 1e-16f);
  alpha[1] = __expf(lrelu02(vs.y + vd.y) - vm.y) / (vq.y + 1e-16f);
  alpha[2] = __expf(lrelu02(vs.z + vd.z) - vm.z) / (vq.z + 1e-16f);
  alpha[3] = __expf(lrelu02(vs.w + vd.w) - vm.w) / (vq.w + 1e-16f);
  const float* hs = &Hsrc[(size_t)s * HIDF];
  float* od = &o[(size_t)d * HIDF];
#pragma unroll
  for (int c = 0; c < HIDF; ++c)
    atomicAdd(&od[c], hs[c] * alpha[c >> 4]);
}

// =============== column-sum of relu(o) -> pool[t*64 + c] ===============
__global__ void pool_relu_kernel(const float* __restrict__ O,
                                 float* __restrict__ pool, int nrows, int tIdx) {
  int c = threadIdx.x;  // 64 threads
  float acc = 0.f;
  for (int n = blockIdx.x; n < nrows; n += gridDim.x) {
    float v = O[(size_t)n * HIDF + c];
    acc += v > 0.f ? v : 0.f;
  }
  atomicAdd(&pool[tIdx * HIDF + c], acc);
}

// =============== semantic score ===============
__global__ void semantic_wmma_kernel(const float* __restrict__ O,
                                     const _Float16* __restrict__ WkPack,
                                     const float* __restrict__ bk,
                                     const float* __restrict__ q,
                                     float* __restrict__ score_sum, int tIdx) {
  const int tile = blockIdx.x;
  const int lane = threadIdx.x;
  const int g = lane >> 4, m = lane & 15;
  const float* rowp = O + (size_t)(tile * 16 + m) * HIDF;
  float4 r0a = *(const float4*)(rowp + 8 * g);
  float4 r0b = *(const float4*)(rowp + 8 * g + 4);
  float4 r1a = *(const float4*)(rowp + 16 + 8 * g);
  float4 r1b = *(const float4*)(rowp + 16 + 8 * g + 4);
  float4 r2a = *(const float4*)(rowp + 32 + 8 * g);
  float4 r2b = *(const float4*)(rowp + 32 + 8 * g + 4);
  float4 r3a = *(const float4*)(rowp + 48 + 8 * g);
  float4 r3b = *(const float4*)(rowp + 48 + 8 * g + 4);
  v16h a0, a1;
  cvt8r(a0, 0, r0a, r0b);  cvt8r(a0, 8, r1a, r1b);
  cvt8r(a1, 0, r2a, r2b);  cvt8r(a1, 8, r3a, r3b);

  float local = 0.f;
#pragma unroll
  for (int nt = 0; nt < 4; ++nt) {
    const int n = nt * 16 + m;
    v16h b0 = *(const v16h*)(WkPack + (size_t)((0 * 4 + nt) * 32 + lane) * 16);
    v16h b1 = *(const v16h*)(WkPack + (size_t)((1 * 4 + nt) * 32 + lane) * 16);
    v8f c;
#pragma unroll
    for (int r = 0; r < 8; ++r) c[r] = bk[n];
    c = __builtin_amdgcn_wmma_f32_16x16x32_f16(false, a0, false, b0, (short)0, c, false, false);
    c = __builtin_amdgcn_wmma_f32_16x16x32_f16(false, a1, false, b1, (short)0, c, false, false);
    float qv = q[n];
#pragma unroll
    for (int r = 0; r < 8; ++r) local += fast_tanh(c[r]) * qv;
  }
#pragma unroll
  for (int off = 16; off > 0; off >>= 1) local += __shfl_down(local, off, 32);
  if (lane == 0) atomicAdd(&score_sum[tIdx], local);
}

// =============== epilogue ===============
__global__ void finalize_kernel(const float* __restrict__ pool,
                                const float* __restrict__ score,
                                const float* __restrict__ cond,
                                const float* __restrict__ eps,
                                const float* __restrict__ W_mu, const float* __restrict__ b_mu,
                                const float* __restrict__ W_lv, const float* __restrict__ b_lv,
                                const float* __restrict__ W_fc3, const float* __restrict__ b_fc3,
                                const float* __restrict__ W_fc2, const float* __restrict__ b_fc2,
                                float* __restrict__ out, int nIng, int nDir) {
  __shared__ float g[136];
  __shared__ float attn[4];
  __shared__ float z[32];
  __shared__ float hdec[64];
  int t = threadIdx.x;  // 64 threads
  if (t == 0) {
    float sc[4] = { score[0] / nDir, score[1] / nDir, score[3] / nDir, score[4] / nDir };
    float mx = fmaxf(fmaxf(sc[0], sc[1]), fmaxf(sc[2], sc[3]));
    float den = 0.f;
    for (int i = 0; i < 4; ++i) { attn[i] = expf(sc[i] - mx); den += attn[i]; }
    for (int i = 0; i < 4; ++i) attn[i] /= den;
  }
  __syncthreads();
  g[t] = pool[2 * 64 + t] / (float)nIng;
  g[64 + t] = (attn[0] * pool[0 * 64 + t] + attn[1] * pool[1 * 64 + t] +
               attn[2] * pool[3 * 64 + t] + attn[3] * pool[4 * 64 + t]) / (float)nDir;
  if (t < 8) g[128 + t] = cond[t];
  __syncthreads();
  if (t < 32) {
    float mu = b_mu[t], lv = b_lv[t];
    for (int i = 0; i < 136; ++i) { mu += g[i] * W_mu[i * 32 + t]; lv += g[i] * W_lv[i * 32 + t]; }
    out[16 + t] = mu;
    out[48 + t] = lv;
    z[t] = mu + eps[t] * expf(0.5f * lv);
  }
  __syncthreads();
  float hv = b_fc3[t];
  for (int i = 0; i < 32; ++i) hv += z[i] * W_fc3[i * 64 + t];
  hdec[t] = fmaxf(hv, 0.f);
  __syncthreads();
  if (t < 16) {
    float r = b_fc2[t];
    for (int i = 0; i < 64; ++i) r += hdec[i] * W_fc2[i * 16 + t];
    out[t] = tanhf(r);
  }
}

extern "C" void kernel_launch(void* const* d_in, const int* in_sizes, int n_in,
                              void* d_out, int out_size, void* d_ws, size_t ws_size,
                              hipStream_t stream) {
  const float* x_ing = (const float*)d_in[0];
  const float* x_dir = (const float*)d_in[1];
  const float* cond  = (const float*)d_in[2];
  const float* eps   = (const float*)d_in[3];
  const int* ei[5] = { (const int*)d_in[4], (const int*)d_in[5], (const int*)d_in[6],
                       (const int*)d_in[7], (const int*)d_in[8] };
  const float* W_pi = (const float*)d_in[9];
  const float* b_pi = (const float*)d_in[10];
  const float* W_pd = (const float*)d_in[11];
  const float* b_pd = (const float*)d_in[12];
  const float* Wk   = (const float*)d_in[13];
  const float* bk   = (const float*)d_in[14];
  const float* q    = (const float*)d_in[15];
  const float* W_mu = (const float*)d_in[16];
  const float* b_mu = (const float*)d_in[17];
  const float* W_lv = (const float*)d_in[18];
  const float* b_lv = (const float*)d_in[19];
  const float* W_fc3 = (const float*)d_in[20];
  const float* b_fc3 = (const float*)d_in[21];
  const float* W_fc2 = (const float*)d_in[22];
  const float* b_fc2 = (const float*)d_in[23];
  const float* att_s[5] = { (const float*)d_in[24], (const float*)d_in[26], (const float*)d_in[28],
                            (const float*)d_in[30], (const float*)d_in[32] };
  const float* att_d[5] = { (const float*)d_in[25], (const float*)d_in[27], (const float*)d_in[29],
                            (const float*)d_in[31], (const float*)d_in[33] };

  const int nIng = in_sizes[0] / 16;
  const int nDir = in_sizes[1] / 8;
  const int E    = in_sizes[4] / 2;

  float* ws = (float*)d_ws;
  float* h_ing = ws;  ws += (size_t)nIng * HIDF;
  float* h_dir = ws;  ws += (size_t)nDir * HIDF;
  float* obuf  = ws;  ws += (size_t)nIng * HIDF;
  float* s_src = ws;  ws += (size_t)nIng * 4;
  float* s_dst = ws;  ws += (size_t)nIng * 4;
  float* amax  = ws;  ws += (size_t)nIng * 4;
  float* denom = ws;  ws += (size_t)nIng * 4;
  _Float16* WpiP = (_Float16*)ws;  ws += 4 * 32 * 16 / 2;
  _Float16* WpdP = (_Float16*)ws;  ws += 4 * 32 * 16 / 2;
  _Float16* WkP  = (_Float16*)ws;  ws += 8 * 32 * 16 / 2;
  float* pool  = ws;  ws += 5 * HIDF;
  float* score = ws;  ws += 8;

  pack_B_kernel<<<1, 128, 0, stream>>>(W_pi, WpiP, 16);
  pack_B_kernel<<<1, 128, 0, stream>>>(W_pd, WpdP, 8);
  pack_Wk_kernel<<<1, 256, 0, stream>>>(Wk, WkP);
  proj_wmma_kernel<16><<<nIng / 16, 32, 0, stream>>>(x_ing, WpiP, b_pi, h_ing);
  proj_wmma_kernel<8> <<<nDir / 16, 32, 0, stream>>>(x_dir, WpdP, b_pd, h_dir);
  zero_kernel<<<1, 512, 0, stream>>>(pool, 5 * HIDF + 8);

  const float* hsT[5] = { h_ing, h_ing, h_dir, h_dir, h_dir };
  const int    nsT[5] = { nIng,  nIng,  nDir,  nDir,  nDir  };
  const float* hdT[5] = { h_dir, h_dir, h_ing, h_dir, h_dir };
  const int    ndT[5] = { nDir,  nDir,  nIng,  nDir,  nDir  };

  const int TB = 256;
  for (int t = 0; t < 5; ++t) {
    const int ns = nsT[t], nd = ndT[t];
    node_scores_kernel<<<(ns * 4 + TB - 1) / TB, TB, 0, stream>>>(hsT[t], att_s[t], s_src, ns * 4);
    node_scores_kernel<<<(nd * 4 + TB - 1) / TB, TB, 0, stream>>>(hdT[t], att_d[t], s_dst, nd * 4);
    seg_init_kernel<<<(nd * HIDF + TB - 1) / TB, TB, 0, stream>>>(obuf, amax, denom, nd);
    edge_amax_kernel<<<(E + TB - 1) / TB, TB, 0, stream>>>(ei[t], E, s_src, s_dst, amax);
    edge_exp_kernel <<<(E + TB - 1) / TB, TB, 0, stream>>>(ei[t], E, s_src, s_dst, amax, denom);
    edge_msg_kernel <<<(E + TB - 1) / TB, TB, 0, stream>>>(ei[t], E, s_src, s_dst, amax, denom,
                                                           hsT[t], obuf);
    pool_relu_kernel<<<256, 64, 0, stream>>>(obuf, pool, nd, t);
    semantic_wmma_kernel<<<nd / 16, 32, 0, stream>>>(obuf, WkP, bk, q, score, t);
  }

  finalize_kernel<<<1, 64, 0, stream>>>(pool, score, cond, eps, W_mu, b_mu, W_lv, b_lv,
                                        W_fc3, b_fc3, W_fc2, b_fc2, (float*)d_out, nIng, nDir);
}
